// Seq2seq__63797444214921
// MI455X (gfx1250) — compile-verified, matrix operand order checked by reading, not run on hardware
//
#include <hip/hip_runtime.h>
#include <hip/hip_bf16.h>
#include <cstdint>
#include <cstddef>

// ---------------------------------------------------------------------------
// Problem constants (match reference)
// ---------------------------------------------------------------------------
static constexpr int NB  = 32;     // batch
static constexpr int NT  = 128;    // time
static constexpr int NV  = 32000;  // vocab
static constexpr int NE  = 1024;   // embed dim
static constexpr int NH  = 1024;   // hidden
static constexpr int N4H = 4096;   // 4*H (gates)
static constexpr int NL  = 3;

typedef __attribute__((ext_vector_type(16))) __bf16 v16bf;
typedef __attribute__((ext_vector_type(8)))  __bf16 v8bf;
typedef __attribute__((ext_vector_type(8)))  float  v8f;

union FragU { struct { v8bf lo, hi; } s; v16bf v; };

// ---------------------------------------------------------------------------
// CDNA5 async global->LDS copy (16 bytes per lane), ASYNCcnt-tracked.
// Builtin param types per hipcc diagnostic: addrspace(1) ptr to
// int __attribute__((vector_size(16))); LDS side is the addrspace(3) analogue.
// ---------------------------------------------------------------------------
typedef int v4i_gcc __attribute__((vector_size(16)));
typedef __attribute__((address_space(1))) v4i_gcc* as1_v4i_ptr;
typedef __attribute__((address_space(3))) v4i_gcc* as3_v4i_ptr;

__device__ __forceinline__ void async_copy_b128(void* lds_ptr, const void* gptr) {
#if __has_builtin(__builtin_amdgcn_global_load_async_to_lds_b128)
  __builtin_amdgcn_global_load_async_to_lds_b128(
      (as1_v4i_ptr)gptr, (as3_v4i_ptr)lds_ptr, 0, 0);
#else
  asm volatile("global_load_async_to_lds_b128 %0, %1, off"
               :: "v"((uint32_t)(uintptr_t)lds_ptr),
                  "v"((uint64_t)(uintptr_t)gptr)
               : "memory");
#endif
}

__device__ __forceinline__ void wait_async0() {
#if __has_builtin(__builtin_amdgcn_s_wait_asynccnt)
  __builtin_amdgcn_s_wait_asynccnt(0);
#else
  asm volatile("s_wait_asynccnt 0" ::: "memory");
#endif
}

// ---------------------------------------------------------------------------
// Convert fp32 [R,C] row-major -> bf16 [C,R] row-major (transpose + downcast).
// ---------------------------------------------------------------------------
__global__ void k_cvt_t(const float* __restrict__ src, __bf16* __restrict__ dst,
                        int R, int C) {
  long i = (long)blockIdx.x * blockDim.x + threadIdx.x;
  long n = (long)R * C;
  if (i >= n) return;
  int r = (int)(i / C), c = (int)(i % C);
  dst[(long)c * R + r] = (__bf16)src[i];
}

// ---------------------------------------------------------------------------
// Embedding gather -> bf16 activations [B*T, E]
// ---------------------------------------------------------------------------
__global__ void k_embed(const int* __restrict__ ids, const float* __restrict__ emb,
                        __bf16* __restrict__ out) {
  long i = (long)blockIdx.x * blockDim.x + threadIdx.x; // NB*NT*NE threads exactly
  int row = (int)(i >> 10);          // / NE
  int e   = (int)(i & (NE - 1));
  out[i] = (__bf16)emb[(long)ids[row] * NE + e];
}

// ---------------------------------------------------------------------------
// Zero the recurrent state (c fp32, h bf16): 32*1024 elements each.
// ---------------------------------------------------------------------------
__global__ void k_zero_state(float* __restrict__ c, __bf16* __restrict__ h) {
  int i = blockIdx.x * 256 + threadIdx.x;   // grid = 128 blocks -> 32768 threads
  c[i] = 0.0f;
  h[i] = (__bf16)0.0f;
}

// ---------------------------------------------------------------------------
// bf16 WMMA GEMM with async-LDS double buffering:
//   C[M,N](f32) = A[M,K](bf16) * Bt[N,K](bf16)^T + bias[N]
// Block = 256 threads (8 waves); block tile 32(M) x 128(N); K-step 32.
// Stage k+1 streams global->LDS via ASYNCcnt-tracked copies while WMMAs
// consume stage k from LDS. Each wave computes 16x32 (2 accumulators,
// one shared A fragment -> 2 WMMAs per A load).
// Staging uses a static transfer->thread map: B tile = 2 copies/thread
// (no branch), A tile = 1 copy for threads 0..127 (wave-aligned scalar
// branch, no EXEC divergence); all addresses precomputed, stage = base + k0.
// Fragment layout per CDNA5 ISA 7.12.2 (16-bit 16x32): lane<16 -> K {0..7,16..23},
// lane>=16 -> K {8..15,24..31}; B mirrors with lane = N column.
// ---------------------------------------------------------------------------
__global__ void k_gemm_bf16(const __bf16* __restrict__ A, const __bf16* __restrict__ Bt,
                            const float* __restrict__ bias, float* __restrict__ C,
                            int M, int N, int K) {
  __shared__ __bf16 As[2][32][32];    //  4 KB
  __shared__ __bf16 Bs[2][128][32];   // 16 KB

  const int tid  = threadIdx.x;
  const int lane = tid & 31;
  const int w    = tid >> 5;
  const int mloc = (w >> 2) * 16;            // 0 / 16
  const int nloc = (w & 3) * 32;             // 0 / 32 / 64 / 96
  const int mblk = blockIdx.y * 32;
  const int nblk = blockIdx.x * 128;

  // ---- static per-thread staging map (all bases hoisted out of K loop) ----
  const int ch    = (tid & 3) * 8;    // 16B chunk within a 64B row-window
  const int bRow1 = tid >> 2;         // B rows 0..63
  const int bRow2 = bRow1 + 64;       // B rows 64..127
  const int aRow  = tid >> 2;         // A rows 0..31 (threads 0..127 only)
  const __bf16* gB1 = Bt + (size_t)(nblk + bRow1) * K + ch;
  const __bf16* gB2 = Bt + (size_t)(nblk + bRow2) * K + ch;
  const __bf16* gA  = A  + (size_t)(mblk + aRow)  * K + ch;

  auto stage = [&](int buf, int k0) {
    async_copy_b128(&Bs[buf][bRow1][ch], gB1 + k0);
    async_copy_b128(&Bs[buf][bRow2][ch], gB2 + k0);
    if (tid < 128)                       // waves 0..3: uniform per-wave branch
      async_copy_b128(&As[buf][aRow][ch], gA + k0);
  };

  // WMMA fragment from an LDS tile (row-major, stride 32)
  auto frag = [&](const __bf16* base, int r0) {
    const __bf16* p = base + (r0 + (lane & 15)) * 32 + ((lane >> 4) << 3);
    FragU u;
    u.s.lo = *(const v8bf*)p;
    u.s.hi = *(const v8bf*)(p + 16);
    return u.v;
  };

  const float bv0 = bias[nblk + nloc + (lane & 15)];
  const float bv1 = bias[nblk + nloc + 16 + (lane & 15)];
  v8f acc0, acc1;
#pragma unroll
  for (int r = 0; r < 8; ++r) { acc0[r] = bv0; acc1[r] = bv1; }

  stage(0, 0);
  wait_async0();
  __syncthreads();

  const int nk = K / 32;
  for (int kt = 0; kt < nk; ++kt) {
    const int buf = kt & 1;
    if (kt + 1 < nk) stage(buf ^ 1, (kt + 1) * 32);  // overlap next stage

    const v16bf a  = frag(&As[buf][0][0], mloc);
    const v16bf b0 = frag(&Bs[buf][0][0], nloc);
    const v16bf b1 = frag(&Bs[buf][0][0], nloc + 16);
    acc0 = __builtin_amdgcn_wmma_f32_16x16x32_bf16(false, a, false, b0,
                                                   (short)0, acc0, false, false);
    acc1 = __builtin_amdgcn_wmma_f32_16x16x32_bf16(false, a, false, b1,
                                                   (short)0, acc1, false, false);
    wait_async0();       // next buffer arrived (and our reads of buf are done)
    __syncthreads();     // block-wide: safe to consume buf^1 / overwrite buf
  }

  const int    col0  = nblk + nloc + (lane & 15);
  const size_t rbase = (size_t)(mblk + mloc + ((lane >> 4) << 3));
#pragma unroll
  for (int r = 0; r < 8; ++r) {
    C[(rbase + r) * N + col0]      = acc0[r];
    C[(rbase + r) * N + col0 + 16] = acc1[r];
  }
}

// ---------------------------------------------------------------------------
// One LSTM timestep: z = Zx[:,t,:] + h_in @ Wh  (WMMA), then gates.
// grid = 64 blocks (16 hidden columns each); block = 256 threads = 8 waves.
// Wave w: gate g = w>>1, batch half mt = w&1 -> one 16x16 z tile, K loop over H.
// All four gate tiles for this column slice meet in LDS -> block-local gate math.
// h double-buffered across launches (every block reads all of h_in).
// Working set (h 64KB + Wh slice 128KB) lives in L2 -> direct global fragments.
// ---------------------------------------------------------------------------
__global__ void k_lstm_step(const float* __restrict__ Zx, const __bf16* __restrict__ WhT,
                            const __bf16* __restrict__ hin, __bf16* __restrict__ hout,
                            float* __restrict__ c, __bf16* __restrict__ outseq, int t) {
  __shared__ float zl[4][32][16];

  const int tid  = threadIdx.x;
  const int lane = tid & 31;
  const int w    = tid >> 5;
  const int g    = w >> 1;        // gate 0..3 (i,f,g,o)
  const int m0   = (w & 1) * 16;  // batch rows 0..15 / 16..31
  const int n0   = blockIdx.x * 16;

  const int ncol = lane & 15;
  const int moff = (lane >> 4) << 3;

  // C init = precomputed x-part (bias already folded in by the Zx GEMM)
  v8f acc;
#pragma unroll
  for (int r = 0; r < 8; ++r)
    acc[r] = Zx[((size_t)((m0 + moff + r) * NT + t)) * N4H + g * NH + n0 + ncol];

  const __bf16* ap = hin + (size_t)(m0 + (lane & 15)) * NH + ((lane >> 4) << 3);
  const __bf16* bp = WhT + (size_t)(g * NH + n0 + (lane & 15)) * NH + ((lane >> 4) << 3);

  for (int k = 0; k < NH; k += 32) {
    if (k + 64 < NH) {                      // uniform branch -> global_prefetch
      __builtin_prefetch(ap + k + 64, 0, 1);
      __builtin_prefetch(bp + k + 64, 0, 1);
    }
    FragU ua, ub;
    ua.s.lo = *(const v8bf*)(ap + k);
    ua.s.hi = *(const v8bf*)(ap + k + 16);
    ub.s.lo = *(const v8bf*)(bp + k);
    ub.s.hi = *(const v8bf*)(bp + k + 16);
    acc = __builtin_amdgcn_wmma_f32_16x16x32_bf16(false, ua.v, false, ub.v,
                                                  (short)0, acc, false, false);
  }

#pragma unroll
  for (int r = 0; r < 8; ++r) zl[g][m0 + moff + r][ncol] = acc[r];
  __syncthreads();

  // 32 batch x 16 cols = 512 elements, 2 per thread
  for (int e = tid; e < 512; e += 256) {
    const int mb = e >> 4, nc = e & 15, col = n0 + nc;
    const float zi = zl[0][mb][nc], zf = zl[1][mb][nc];
    const float zg = zl[2][mb][nc], zo = zl[3][mb][nc];
    const float ig = 1.0f / (1.0f + __expf(-zi));
    const float fg = 1.0f / (1.0f + __expf(-zf));
    const float gg = tanhf(zg);
    const float og = 1.0f / (1.0f + __expf(-zo));
    const float cn = fg * c[mb * NH + col] + ig * gg;
    const float hn = og * tanhf(cn);
    c[mb * NH + col] = cn;
    const __bf16 hb = (__bf16)hn;
    hout[mb * NH + col] = hb;
    outseq[((size_t)(mb * NT + t)) * NH + col] = hb;
  }
}

// ---------------------------------------------------------------------------
// Host-side orchestration
// ---------------------------------------------------------------------------
extern "C" void kernel_launch(void* const* d_in, const int* in_sizes, int n_in,
                              void* d_out, int out_size, void* d_ws, size_t ws_size,
                              hipStream_t stream) {
  (void)in_sizes; (void)n_in; (void)out_size; (void)ws_size;

  const int*   enc_ids = (const int*)d_in[0];
  const int*   dec_ids = (const int*)d_in[1];
  const float* emb     = (const float*)d_in[2];
  const float* Wx_enc  = (const float*)d_in[3];
  const float* Wh_enc  = (const float*)d_in[4];
  const float* b_enc   = (const float*)d_in[5];
  const float* Wx_dec  = (const float*)d_in[6];
  const float* Wh_dec  = (const float*)d_in[7];
  const float* b_dec   = (const float*)d_in[8];
  const float* dW      = (const float*)d_in[9];
  const float* db      = (const float*)d_in[10];
  float* out = (float*)d_out;

  // ---- carve workspace ----
  uint8_t* ws = (uint8_t*)d_ws;
  size_t off = 0;
  auto carve = [&](size_t bytes) -> void* {
    off = (off + 255) & ~(size_t)255;
    void* p = ws + off;
    off += bytes;
    return p;
  };
  __bf16* WxT[2 * NL];
  __bf16* WhT[2 * NL];
  for (int l = 0; l < 2 * NL; ++l) {
    WxT[l] = (__bf16*)carve((size_t)N4H * NE * 2);
    WhT[l] = (__bf16*)carve((size_t)N4H * NH * 2);
  }
  __bf16* WdT  = (__bf16*)carve((size_t)NV * NH * 2);
  __bf16* bufA = (__bf16*)carve((size_t)NB * NT * NH * 2);
  __bf16* bufB = (__bf16*)carve((size_t)NB * NT * NH * 2);
  float*  Zx   = (float*) carve((size_t)NB * NT * N4H * 4);
  float*  cst  = (float*) carve((size_t)NB * NH * 4);
  __bf16* hA   = (__bf16*)carve((size_t)NB * NH * 2);
  __bf16* hB   = (__bf16*)carve((size_t)NB * NH * 2);

  // ---- weight conversion (fp32 -> bf16, transposed to [N,K]) ----
  const int cvtBlocks = (NE * N4H) / 256;  // 16384
  for (int l = 0; l < NL; ++l) {
    k_cvt_t<<<cvtBlocks, 256, 0, stream>>>(Wx_enc + (size_t)l * NE * N4H, WxT[l],      NE, N4H);
    k_cvt_t<<<cvtBlocks, 256, 0, stream>>>(Wh_enc + (size_t)l * NH * N4H, WhT[l],      NH, N4H);
    k_cvt_t<<<cvtBlocks, 256, 0, stream>>>(Wx_dec + (size_t)l * NE * N4H, WxT[NL + l], NE, N4H);
    k_cvt_t<<<cvtBlocks, 256, 0, stream>>>(Wh_dec + (size_t)l * NH * N4H, WhT[NL + l], NH, N4H);
  }
  k_cvt_t<<<(NH * NV) / 256, 256, 0, stream>>>(dW, WdT, NH, NV);

  __bf16* cur = bufA;
  __bf16* nxt = bufB;

  // ---- encoder: 3 stacked LSTMs, zero initial state each ----
  k_embed<<<(NB * NT * NE) / 256, 256, 0, stream>>>(enc_ids, emb, cur);
  for (int l = 0; l < NL; ++l) {
    k_zero_state<<<128, 256, 0, stream>>>(cst, hA);
    k_gemm_bf16<<<dim3(N4H / 128, (NB * NT) / 32), 256, 0, stream>>>(
        cur, WxT[l], b_enc + (size_t)l * N4H, Zx, NB * NT, N4H, NE);
    for (int t = 0; t < NT; ++t) {
      __bf16* hr = (t & 1) ? hB : hA;
      __bf16* hw = (t & 1) ? hA : hB;
      k_lstm_step<<<NH / 16, 256, 0, stream>>>(Zx, WhT[l], hr, hw, cst, nxt, t);
    }
    __bf16* tmp = cur; cur = nxt; nxt = tmp;  // NT even -> final state lands in hA
  }

  // ---- decoder: state chains from encoder final, then layer-to-layer ----
  k_embed<<<(NB * NT * NE) / 256, 256, 0, stream>>>(dec_ids, emb, cur);
  for (int l = 0; l < NL; ++l) {
    // no zeroing: hA/cst already hold the chained initial state
    k_gemm_bf16<<<dim3(N4H / 128, (NB * NT) / 32), 256, 0, stream>>>(
        cur, WxT[NL + l], b_dec + (size_t)l * N4H, Zx, NB * NT, N4H, NE);
    for (int t = 0; t < NT; ++t) {
      __bf16* hr = (t & 1) ? hB : hA;
      __bf16* hw = (t & 1) ? hA : hB;
      k_lstm_step<<<NH / 16, 256, 0, stream>>>(Zx, WhT[NL + l], hr, hw, cst, nxt, t);
    }
    __bf16* tmp = cur; cur = nxt; nxt = tmp;
  }

  // ---- dense head: logits[B*T, V] = Y @ dense_W + dense_b ----
  k_gemm_bf16<<<dim3(NV / 128, (NB * NT) / 32), 256, 0, stream>>>(
      cur, WdT, db, out, NB * NT, NV, NH);
}
